// Attention_22986664968925
// MI455X (gfx1250) — matrix operand order repaired and checked by reading, not compiled, not run on hardware
//
#include <hip/hip_runtime.h>
#include <hip/hip_bf16.h>
#include <math.h>

// ---------------------------------------------------------------------------
// MI455X (gfx1250) attention layer, bf16 WMMA path (v_wmma_f32_16x16x32_bf16)
// ---------------------------------------------------------------------------

typedef __attribute__((ext_vector_type(16))) __bf16 v16bf;
typedef __attribute__((ext_vector_type(8)))  __bf16 v8bf;
typedef __attribute__((ext_vector_type(8)))  float  v8f;

#define D_MODEL 1024
#define SEQ     2048
#define BATCH   8
#define ROWS_PER_SRC (BATCH * SEQ)      // 16384
#define ATTN_SCALE (0.03125f)           // 1/sqrt(1024)
#define NEG_INF_F (-1.0e9f)
#define SPITCH  (SEQ + 8)               // f32 score pitch: 8-dword row rotation
#define BPITCH  (SEQ + 16)              // bf16 prob pitch: 4128B rows -> 8-bank rotation

// ---- WMMA fragment helpers (layouts per CDNA5 ISA 7.12.2) ------------------

// A matrix 16x32 bf16, row-major source. Lane holds row M = lane%16.
// element e <-> k = half*16 + e: one contiguous 32B run.
__device__ __forceinline__ v16bf fragA_bf16(const __bf16* rk, int half) {
  v8bf lo = *(const v8bf*)(rk + half * 16);
  v8bf hi = *(const v8bf*)(rk + half * 16 + 8);
  return __builtin_shufflevector(lo, hi, 0,1,2,3,4,5,6,7,8,9,10,11,12,13,14,15);
}

// B matrix 32x16 bf16 sourced from row n (= lane%16) of a [N][K] array:
// lane holds column N; element e <-> k = half*16 + e: one contiguous 32B run.
__device__ __forceinline__ v16bf fragB_bf16(const __bf16* rk, int half) {
  v8bf lo = *(const v8bf*)(rk + half * 16);
  v8bf hi = *(const v8bf*)(rk + half * 16 + 8);
  return __builtin_shufflevector(lo, hi, 0,1,2,3,4,5,6,7,8,9,10,11,12,13,14,15);
}

__device__ __forceinline__ v8f wmma_bf16(v16bf a, v16bf b, v8f c) {
  // (neg_a, A, neg_b, B, c_mod, C, reuse_a, reuse_b)
  return __builtin_amdgcn_wmma_f32_16x16x32_bf16(false, a, false, b, (short)0, c,
                                                 false, false);
}

// ---- Kernel 0: f32 -> bf16 conversion (weights and q|k|v activations) ------

__global__ void cvt_f32_bf16(const float* __restrict__ src,
                             __bf16* __restrict__ dst, int n) {
  int i = (blockIdx.x * blockDim.x + threadIdx.x) * 8;
  if (i + 8 <= n) {
#pragma unroll
    for (int j = 0; j < 8; ++j) dst[i + j] = (__bf16)src[i + j];
  }
}

// ---- Kernel 1: fused QKV projection  (qp/kp row-major bf16, vp transposed) --
// Block = 8 waves; wave computes a 16(M) x 64(N) strip (4 accumulators).
// Grid: x = 49152/128 M-tiles (q,k,v stacked), y = 1024/64 N-tiles.
// A is pre-converted bf16, so the inner loop is pure b128 loads + WMMA.

__global__ __launch_bounds__(256)
void proj_kernel(const __bf16* __restrict__ qkvb, const __bf16* __restrict__ wqb,
                 const float* __restrict__ bq,
                 __bf16* __restrict__ qp, __bf16* __restrict__ kp,
                 __bf16* __restrict__ vpT) {
  const int wave = threadIdx.x >> 5;
  const int lane = threadIdx.x & 31;
  const int half = lane >> 4;
  const int l16  = lane & 15;

  const int rowBlock  = blockIdx.x * 128;            // [0, 49152)
  const int src       = rowBlock / ROWS_PER_SRC;     // 0=q 1=k 2=v (uniform per block)
  const int rowInSrc0 = rowBlock - src * ROWS_PER_SRC + wave * 16;
  const int colBase   = blockIdx.y * 64;

  const __bf16* arow = qkvb + (size_t)(rowBlock + wave * 16 + l16) * D_MODEL;

  v8f acc[4] = {};
  for (int kb = 0; kb < D_MODEL; kb += 32) {
    v16bf af = fragA_bf16(arow + kb, half);
#pragma unroll
    for (int j = 0; j < 4; ++j) {
      const __bf16* brow = wqb + (size_t)(colBase + j * 16 + l16) * D_MODEL + kb;
      acc[j] = wmma_bf16(af, fragB_bf16(brow, half), acc[j]);
    }
  }

#pragma unroll
  for (int j = 0; j < 4; ++j) {
    const int col   = colBase + j * 16 + l16;
    const float bias = bq[col];
    if (src == 2) {
      // store transposed: vpT[b][d][s]; the 8 C rows are contiguous in s.
      const int rows0 = rowInSrc0 + half * 8;
      const int b  = rows0 / SEQ;
      const int s  = rows0 - b * SEQ;
      v8bf pk;
#pragma unroll
      for (int r = 0; r < 8; ++r) pk[r] = (__bf16)(acc[j][r] + bias);
      *(v8bf*)(vpT + ((size_t)b * D_MODEL + col) * SEQ + s) = pk;
    } else {
      __bf16* dst = (src == 0) ? qp : kp;
#pragma unroll
      for (int r = 0; r < 8; ++r)
        dst[(size_t)(rowInSrc0 + half * 8 + r) * D_MODEL + col] =
            (__bf16)(acc[j][r] + bias);
    }
  }
}

// ---- Kernel 2: attention core ----------------------------------------------
// One workgroup = (batch b, 16 query rows). Per-batch K/V (8 MB bf16) stays
// L2-resident across the 128 q-tiles, so re-reads never touch HBM.
// Phase 1: S = qp*kp^T via WMMA -> padded LDS (f32, 16 x 2048).
// Phase 2: exact masked softmax; writes weights (f32, coalesced) + normalized
//          bf16 probabilities into a second LDS region.
// Phase 3: ctx = P*vp via WMMA; A frags come straight from LDS bf16 (no cvt).

__global__ __launch_bounds__(256)
void attn_kernel(const __bf16* __restrict__ qp, const __bf16* __restrict__ kp,
                 const __bf16* __restrict__ vpT, const int* __restrict__ mask,
                 float* __restrict__ weights, __bf16* __restrict__ ctx) {
  extern __shared__ char smem[];
  float*  sc = (float*)smem;                           // [16][SPITCH] f32 scores
  __bf16* pb = (__bf16*)(smem + 16 * SPITCH * 4);      // [16][BPITCH] bf16 probs

  const int wave = threadIdx.x >> 5;
  const int lane = threadIdx.x & 31;
  const int half = lane >> 4;
  const int l16  = lane & 15;
  const int b     = blockIdx.x >> 7;                   // /128
  const int qt    = blockIdx.x & 127;
  const int qbase = qt * 16;

  // --- Phase 1: scores -------------------------------------------------------
  const __bf16* qrow = qp + (size_t)(b * SEQ + qbase + l16) * D_MODEL;
  for (int ct = wave; ct < SEQ / 16; ct += 8) {
    const __bf16* krow = kp + (size_t)(b * SEQ + ct * 16 + l16) * D_MODEL;
    if (ct + 8 < SEQ / 16) {
      // pull the next K tile's row toward L1 (global_prefetch_b8)
      __builtin_prefetch(krow + 8 * 16 * D_MODEL, 0, 1);
    }
    v8f acc = {};
#pragma unroll 4
    for (int kb = 0; kb < D_MODEL; kb += 32) {
      acc = wmma_bf16(fragA_bf16(qrow + kb, half), fragB_bf16(krow + kb, half), acc);
    }
#pragma unroll
    for (int r = 0; r < 8; ++r)
      sc[(half * 8 + r) * SPITCH + ct * 16 + l16] = acc[r];
  }
  __syncthreads();

  // --- Phase 2: masked softmax ----------------------------------------------
  const int* mrow = mask + b * SEQ;
  float* wg = weights + ((size_t)(b * SEQ + qbase)) * SEQ;
#pragma unroll
  for (int rr = 0; rr < 2; ++rr) {
    const int row = wave * 2 + rr;
    float*  srow = sc + row * SPITCH;
    __bf16* prow = pb + row * BPITCH;
    float mx = NEG_INF_F;
#pragma unroll 4
    for (int j = 0; j < 64; ++j) {
      const int kk = lane + 32 * j;
      float s = srow[kk] * ATTN_SCALE;
      s = (mrow[kk] == 0) ? NEG_INF_F : s;
      srow[kk] = s;
      mx = fmaxf(mx, s);
    }
#pragma unroll
    for (int off = 16; off > 0; off >>= 1) mx = fmaxf(mx, __shfl_xor(mx, off, 32));
    float sum = 0.f;
#pragma unroll 4
    for (int j = 0; j < 64; ++j) {
      const int kk = lane + 32 * j;
      float p = __expf(srow[kk] - mx);
      srow[kk] = p;
      sum += p;
    }
#pragma unroll
    for (int off = 16; off > 0; off >>= 1) sum += __shfl_xor(sum, off, 32);
    const float inv = 1.0f / sum;
#pragma unroll 4
    for (int j = 0; j < 64; ++j) {
      const int kk = lane + 32 * j;
      const float p = srow[kk] * inv;
      prow[kk] = (__bf16)p;                          // bf16 prob for P@V
      wg[(size_t)row * SEQ + kk] = p;                // full weights output (f32)
    }
  }
  __syncthreads();

  // --- Phase 3: ctx = P @ vp ------------------------------------------------
  for (int ct = wave; ct < D_MODEL / 16; ct += 8) {
    const int dcol = ct * 16 + l16;
    const __bf16* vrow = vpT + ((size_t)b * D_MODEL + dcol) * SEQ;
    if (ct + 8 < D_MODEL / 16) {
      __builtin_prefetch(vrow + (size_t)8 * 16 * SEQ, 0, 1);
    }
    v8f acc = {};
#pragma unroll 4
    for (int kb = 0; kb < SEQ; kb += 32) {
      acc = wmma_bf16(fragA_bf16(pb + l16 * BPITCH + kb, half),
                      fragB_bf16(vrow + kb, half), acc);
    }
#pragma unroll
    for (int r = 0; r < 8; ++r)
      ctx[(size_t)(b * SEQ + qbase + half * 8 + r) * D_MODEL + dcol] =
          (__bf16)acc[r];
  }
}

// ---- Kernel 3: output projection (f32 epilogue + bias) ---------------------

__global__ __launch_bounds__(256)
void outproj_kernel(const __bf16* __restrict__ ctx, const __bf16* __restrict__ wob,
                    const float* __restrict__ bout, float* __restrict__ out) {
  const int wave = threadIdx.x >> 5;
  const int lane = threadIdx.x & 31;
  const int half = lane >> 4;
  const int l16  = lane & 15;
  const int row0    = blockIdx.x * 128 + wave * 16;
  const int colBase = blockIdx.y * 64;

  const __bf16* arow = ctx + (size_t)(row0 + l16) * D_MODEL;
  v8f acc[4] = {};
  for (int kb = 0; kb < D_MODEL; kb += 32) {
    v16bf af = fragA_bf16(arow + kb, half);
#pragma unroll
    for (int j = 0; j < 4; ++j) {
      const __bf16* brow = wob + (size_t)(colBase + j * 16 + l16) * D_MODEL + kb;
      acc[j] = wmma_bf16(af, fragB_bf16(brow, half), acc[j]);
    }
  }
#pragma unroll
  for (int j = 0; j < 4; ++j) {
    const int col = colBase + j * 16 + l16;
    const float bias = bout[col];
#pragma unroll
    for (int r = 0; r < 8; ++r)
      out[(size_t)(row0 + half * 8 + r) * D_MODEL + col] = acc[j][r] + bias;
  }
}

// ---- Host launcher ----------------------------------------------------------

extern "C" void kernel_launch(void* const* d_in, const int* in_sizes, int n_in,
                              void* d_out, int out_size, void* d_ws, size_t ws_size,
                              hipStream_t stream) {
  (void)in_sizes; (void)n_in; (void)out_size; (void)ws_size;

  const float* q    = (const float*)d_in[0];
  const float* k    = (const float*)d_in[1];
  const float* v    = (const float*)d_in[2];
  const int*   mask = (const int*)d_in[3];
  const float* Wq   = (const float*)d_in[4];
  const float* bq   = (const float*)d_in[5];
  const float* Wout = (const float*)d_in[6];
  const float* bout = (const float*)d_in[7];

  // Workspace (bf16): Wq(2MB) Wout(2MB) qkv(96MB) qp(32MB) kp(32MB) vpT(32MB)
  // ctx(32MB)  => ~228MB total
  __bf16* wqb  = (__bf16*)d_ws;
  __bf16* wob  = wqb  + (size_t)D_MODEL * D_MODEL;
  __bf16* qkvb = wob  + (size_t)D_MODEL * D_MODEL;
  __bf16* qp   = qkvb + (size_t)3 * ROWS_PER_SRC * D_MODEL;
  __bf16* kp   = qp   + (size_t)ROWS_PER_SRC * D_MODEL;
  __bf16* vpT  = kp   + (size_t)ROWS_PER_SRC * D_MODEL;
  __bf16* ctx  = vpT  + (size_t)ROWS_PER_SRC * D_MODEL;

  float* out     = (float*)d_out;
  float* weights = out + (size_t)ROWS_PER_SRC * D_MODEL;

  const int nW = D_MODEL * D_MODEL;              // 1M
  const int nA = ROWS_PER_SRC * D_MODEL;         // 16.7M
  cvt_f32_bf16<<<nW / 2048, 256, 0, stream>>>(Wq,   wqb, nW);
  cvt_f32_bf16<<<nW / 2048, 256, 0, stream>>>(Wout, wob, nW);
  cvt_f32_bf16<<<nA / 2048, 256, 0, stream>>>(q, qkvb,              nA);
  cvt_f32_bf16<<<nA / 2048, 256, 0, stream>>>(k, qkvb + (size_t)nA, nA);
  cvt_f32_bf16<<<nA / 2048, 256, 0, stream>>>(v, qkvb + 2*(size_t)nA, nA);

  proj_kernel<<<dim3(3 * ROWS_PER_SRC / 128, D_MODEL / 64), 256, 0, stream>>>(
      qkvb, wqb, bq, qp, kp, vpT);

  const size_t smem = (size_t)16 * SPITCH * 4 + (size_t)16 * BPITCH * 2; // ~193KB
  hipFuncSetAttribute((const void*)attn_kernel,
                      hipFuncAttributeMaxDynamicSharedMemorySize, (int)smem);
  attn_kernel<<<dim3(BATCH * (SEQ / 16)), 256, smem, stream>>>(
      qp, kp, vpT, mask, weights, ctx);

  outproj_kernel<<<dim3(ROWS_PER_SRC / 128, D_MODEL / 64), 256, 0, stream>>>(
      ctx, wob, bout, out);
}
